// Adapter2_88476326297964
// MI455X (gfx1250) — compile-verified, matrix operand order checked by reading, not run on hardware
//
#include <hip/hip_runtime.h>

// ---------------------------------------------------------------------------
// Adapter2 fused implementation for gfx1250 (MI455X).
//   C=768, CA=192, L=197, N=256 (B=16*T=16), M = L*N = 50432.
// GEMM1: Xb(50432x768 bf16) @ [fc1|mlp_in|off_fc1](768x576) -> wmma bf16/f32
// mid:   temporal dw-conv | quickgelu (in GEMM1 epilogue) | shift-diff + 3x3 dw
// GEMM2: U2(50432x576) @ [fc2;mlp_out;off_fc2](576x768) -> f32 out (summed)
// GEMM pipeline: 256x64 block tile, 8 waves (32M x 64N each, 2x4 wmma tiles),
// double-buffered LDS fed by GLOBAL_LOAD_ASYNC_TO_LDS_B128 (ASYNCcnt).
// ---------------------------------------------------------------------------

typedef __attribute__((ext_vector_type(16))) __bf16 v16bf;
typedef __attribute__((ext_vector_type(8)))  float  v8f;
typedef __attribute__((ext_vector_type(4)))  int    v4i;

union Frag { v16bf v; uint4 q[2]; };

#define AS1 __attribute__((address_space(1)))
#define AS3 __attribute__((address_space(3)))

#if __has_builtin(__builtin_amdgcn_global_load_async_to_lds_b128) && \
    __has_builtin(__builtin_amdgcn_s_wait_asynccnt)
#define USE_ASYNC_LDS 1
#else
#define USE_ASYNC_LDS 0
#endif

__device__ __forceinline__ void cp_b128(const unsigned short* g, unsigned short* l) {
#if USE_ASYNC_LDS
  __builtin_amdgcn_global_load_async_to_lds_b128(
      (AS1 v4i*)(AS1 void*)(void*)g,
      (AS3 v4i*)(AS3 void*)(void*)l, 0, 0);
#else
  *reinterpret_cast<uint4*>(l) = *reinterpret_cast<const uint4*>(g);
#endif
}
__device__ __forceinline__ void wait_async() {
#if USE_ASYNC_LDS
  __builtin_amdgcn_s_wait_asynccnt(0);
#endif
}

__device__ __forceinline__ unsigned short f2bf(float f) {
  unsigned u = __float_as_uint(f);
  unsigned r = u + 0x7FFFu + ((u >> 16) & 1u);   // round-to-nearest-even
  return (unsigned short)(r >> 16);
}
__device__ __forceinline__ float bf2f(unsigned short h) {
  return __uint_as_float(((unsigned)h) << 16);
}

static constexpr int Cc   = 768;
static constexpr int CA   = 192;
static constexpr int Lc   = 197;
static constexpr int Nc   = 256;
static constexpr int Mtot = Lc * Nc;       // 50432 = 197 * 256
static constexpr int K1   = 768;
static constexpr int N1   = 576;           // 3 * 192
static constexpr int K2   = 576;
static constexpr int N2   = 768;

// --------------------- X fp32 -> bf16 (one pass) ----------------------------
__global__ __launch_bounds__(256) void convert_x(
    const float* __restrict__ X, unsigned short* __restrict__ Xb)
{
  size_t i = ((size_t)blockIdx.x * 256 + threadIdx.x) * 8;   // 8 elems/thread
  const float4* p = reinterpret_cast<const float4*>(X + i);
  float4 a = p[0], b = p[1];
  uint4 o;
  o.x = (unsigned)f2bf(a.x) | ((unsigned)f2bf(a.y) << 16);
  o.y = (unsigned)f2bf(a.z) | ((unsigned)f2bf(a.w) << 16);
  o.z = (unsigned)f2bf(b.x) | ((unsigned)f2bf(b.y) << 16);
  o.w = (unsigned)f2bf(b.z) | ((unsigned)f2bf(b.w) << 16);
  *reinterpret_cast<uint4*>(Xb + i) = o;
}

// --------------------------- weight prep ------------------------------------
// W1t[n][k] (576x768 bf16), W2t[c][k] (768x576 bf16): B^T so fragment slices
// are contiguous 16B chunks.
__global__ __launch_bounds__(256) void prep_weights(
    const float* __restrict__ fc1_w, const float* __restrict__ mlp_in_w,
    const float* __restrict__ off_fc1_w,
    const float* __restrict__ fc2_w, const float* __restrict__ mlp_out_w,
    const float* __restrict__ off_fc2_w,
    unsigned short* __restrict__ W1t, unsigned short* __restrict__ W2t)
{
  int idx = blockIdx.x * 256 + threadIdx.x;
  const int T1 = N1 * K1;
  if (idx < T1) {
    int nn = idx / K1, k = idx % K1;
    int s = nn / CA, cc = nn % CA;
    const float* src = (s == 0) ? fc1_w : (s == 1) ? mlp_in_w : off_fc1_w;
    W1t[idx] = f2bf(src[k * CA + cc]);                 // (C,CA) row-major
  } else {
    int j = idx - T1;
    if (j < N2 * K2) {
      int c = j / K2, k = j % K2;
      int s = k / CA, cc = k % CA;
      const float* src = (s == 0) ? fc2_w : (s == 1) ? mlp_out_w : off_fc2_w;
      W2t[j] = f2bf(src[cc * Cc + c]);                 // (CA,C) row-major
    }
  }
}

// ----------------- shared WMMA mainloop (256M x 64N block) ------------------
// 8 waves stacked in M (wave w: rows [w*32, w*32+32)), each wave 2x4 tiles.
template<int KD>
__device__ __forceinline__ void gemm_tile(
    const unsigned short* __restrict__ A,   // (Mtot x KD) bf16 row-major
    const unsigned short* __restrict__ B,   // (Ntot x KD) bf16 row-major (B^T)
    int mBase, int nBase, v8f (&acc)[2][4])
{
  __shared__ unsigned short As[2][256][40];   // 80B row stride (16B aligned)
  __shared__ unsigned short Bs[2][64][40];

  const int tid  = threadIdx.x;
  const int lane = tid & 31;
  const int w    = tid >> 5;
  const int mr   = lane & 15;
  const int kb   = (lane >> 4) * 8;     // ISA 16-bit A/B lane K layout

  const unsigned short* aRow = A + (size_t)(mBase + tid) * KD;
  const unsigned short* bRow = B + (size_t)(nBase + (tid & 63)) * KD + (tid >> 6) * 8;
  unsigned short* aDst0 = &As[0][tid][0];
  unsigned short* aDst1 = &As[1][tid][0];
  unsigned short* bDst0 = &Bs[0][tid & 63][(tid >> 6) * 8];
  unsigned short* bDst1 = &Bs[1][tid & 63][(tid >> 6) * 8];

  auto stage = [&](int buf, int kt) {
    unsigned short* ad = buf ? aDst1 : aDst0;
    #pragma unroll
    for (int c = 0; c < 4; ++c)
      cp_b128(aRow + kt + c * 8, ad + c * 8);
    cp_b128(bRow + kt, buf ? bDst1 : bDst0);
  };

  stage(0, 0);
  int buf = 0;
  for (int kt = 0; kt < KD; kt += 32) {
    wait_async();          // this wave's async copies for `buf` have landed
    __syncthreads();       // everyone's have landed; prev compute done
    if (kt + 32 < KD) stage(buf ^ 1, kt + 32);

    Frag fa[2], fb[4];
    #pragma unroll
    for (int i = 0; i < 2; ++i) {
      const unsigned short* p = &As[buf][w * 32 + i * 16 + mr][kb];
      fa[i].q[0] = *reinterpret_cast<const uint4*>(p);
      fa[i].q[1] = *reinterpret_cast<const uint4*>(p + 16);
    }
    #pragma unroll
    for (int j = 0; j < 4; ++j) {
      const unsigned short* p = &Bs[buf][j * 16 + mr][kb];
      fb[j].q[0] = *reinterpret_cast<const uint4*>(p);
      fb[j].q[1] = *reinterpret_cast<const uint4*>(p + 16);
    }
    #pragma unroll
    for (int i = 0; i < 2; ++i)
      #pragma unroll
      for (int j = 0; j < 4; ++j)
        acc[i][j] = __builtin_amdgcn_wmma_f32_16x16x32_bf16(
            false, fa[i].v, false, fb[j].v, (short)0, acc[i][j], false, false);
    buf ^= 1;
  }
}

// ------------------------------ GEMM1 ---------------------------------------
__global__ __launch_bounds__(256) void gemm1_wmma(
    const unsigned short* __restrict__ Xb, const unsigned short* __restrict__ W1t,
    const float* __restrict__ fc1_b, const float* __restrict__ mlp_in_b,
    unsigned short* __restrict__ Utmp, unsigned short* __restrict__ U2)
{
  v8f acc[2][4];
  #pragma unroll
  for (int i = 0; i < 2; ++i)
    #pragma unroll
    for (int j = 0; j < 4; ++j)
      acc[i][j] = v8f{0.f,0.f,0.f,0.f,0.f,0.f,0.f,0.f};

  const int mBase = blockIdx.y * 256;
  const int nBase = blockIdx.x * 64;
  gemm_tile<K1>(Xb, W1t, mBase, nBase, acc);

  const int lane = threadIdx.x & 31;
  const int w    = threadIdx.x >> 5;
  const int rB   = mBase + w * 32 + ((lane >> 4) << 3);
  const int sec  = nBase / CA;                 // uniform per block (64 | 192)

  if (sec == 0) {                              // main h1 (pre temporal conv)
    #pragma unroll
    for (int j = 0; j < 4; ++j) {
      const int cc = nBase + j * 16 + (lane & 15);
      const float bias = fc1_b[cc];
      #pragma unroll
      for (int i = 0; i < 2; ++i)
        #pragma unroll
        for (int r = 0; r < 8; ++r)
          Utmp[(size_t)(rB + i * 16 + r) * 384 + cc] = f2bf(acc[i][j][r] + bias);
    }
  } else if (sec == 1) {                       // mlp: finish quickgelu here
    #pragma unroll
    for (int j = 0; j < 4; ++j) {
      const int cc = nBase - CA + j * 16 + (lane & 15);
      const float bias = mlp_in_b[cc];
      #pragma unroll
      for (int i = 0; i < 2; ++i)
        #pragma unroll
        for (int r = 0; r < 8; ++r) {
          float v = acc[i][j][r] + bias;
          float s = 1.0f / (1.0f + __expf(-1.702f * v));
          U2[(size_t)(rB + i * 16 + r) * K2 + CA + cc] = f2bf(v * s);
        }
    }
  } else {                                     // y_off = x @ off_fc1 (no bias)
    #pragma unroll
    for (int j = 0; j < 4; ++j) {
      const int cc = nBase - 2 * CA + j * 16 + (lane & 15);
      #pragma unroll
      for (int i = 0; i < 2; ++i)
        #pragma unroll
        for (int r = 0; r < 8; ++r)
          Utmp[(size_t)(rB + i * 16 + r) * 384 + CA + cc] = f2bf(acc[i][j][r]);
    }
  }
}

// --------------------------- mid stage (convs) -------------------------------
__global__ __launch_bounds__(256) void stage2_conv(
    const unsigned short* __restrict__ Utmp,
    const float* __restrict__ conv_w, const float* __restrict__ conv_b,
    const float* __restrict__ off_fc1_b,
    const float* __restrict__ off_conv_w, const float* __restrict__ off_conv_b,
    unsigned short* __restrict__ U2)
{
  int idx = blockIdx.x * 256 + threadIdx.x;
  int ca = idx % CA;
  int m  = idx / CA;
  if (m >= Mtot) return;
  int l = m >> 8;
  int n = m & 255;
  int b = n >> 4, t = n & 15;

  // main branch: depthwise conv1d over T (zero pad)
  float s = conv_b[ca];
  #pragma unroll
  for (int dt = -1; dt <= 1; ++dt) {
    int tt = t + dt;
    if (tt >= 0 && tt < 16)
      s += conv_w[ca * 3 + dt + 1] *
           bf2f(Utmp[(size_t)(l * Nc + b * 16 + tt) * 384 + ca]);
  }
  U2[(size_t)m * K2 + ca] = f2bf(s);

  // offset branch: diff@W + bias == y[t]-y[former] + off_fc1_b, then 3x3 dw.
  float o = 0.0f;
  if (l > 0) {
    int hw = l - 1;
    int h = hw / 14, w = hw % 14;
    int nf = b * 16 + (t > 0 ? t - 1 : 0);
    float ob = off_fc1_b[ca];
    o = off_conv_b[ca];
    #pragma unroll
    for (int dh = 0; dh < 3; ++dh) {
      int hh = h + dh - 1;
      if (hh < 0 || hh >= 14) continue;
      #pragma unroll
      for (int dw = 0; dw < 3; ++dw) {
        int ww = w + dw - 1;
        if (ww < 0 || ww >= 14) continue;
        size_t base = (size_t)(1 + hh * 14 + ww) * Nc;
        float ya = bf2f(Utmp[(base + n ) * 384 + CA + ca]);
        float yb = bf2f(Utmp[(base + nf) * 384 + CA + ca]);
        o += off_conv_w[ca * 9 + dh * 3 + dw] * (ya - yb + ob);
      }
    }
  }
  U2[(size_t)m * K2 + 2 * CA + ca] = f2bf(o);  // zero for CLS row (l==0)
}

// ------------------------------ GEMM2 ---------------------------------------
__global__ __launch_bounds__(256) void gemm2_wmma(
    const unsigned short* __restrict__ U2, const unsigned short* __restrict__ W2t,
    const float* __restrict__ fc2_b, const float* __restrict__ mlp_out_b,
    const float* __restrict__ off_fc2_b, float* __restrict__ out)
{
  v8f acc[2][4];
  #pragma unroll
  for (int i = 0; i < 2; ++i)
    #pragma unroll
    for (int j = 0; j < 4; ++j)
      acc[i][j] = v8f{0.f,0.f,0.f,0.f,0.f,0.f,0.f,0.f};

  const int mBase = blockIdx.y * 256;
  const int nBase = blockIdx.x * 64;
  gemm_tile<K2>(U2, W2t, mBase, nBase, acc);

  const int lane = threadIdx.x & 31;
  const int w    = threadIdx.x >> 5;
  const int rB   = mBase + w * 32 + ((lane >> 4) << 3);
  const bool addOff = (mBase != 0);   // rows 0..255 are exactly the CLS rows

  #pragma unroll
  for (int j = 0; j < 4; ++j) {
    const int col = nBase + j * 16 + (lane & 15);
    float bias = fc2_b[col] + mlp_out_b[col];
    if (addOff) bias += off_fc2_b[col];
    #pragma unroll
    for (int i = 0; i < 2; ++i)
      #pragma unroll
      for (int r = 0; r < 8; ++r)
        out[(size_t)(rB + i * 16 + r) * Cc + col] = acc[i][j][r] + bias;
  }
}

// ------------------------------ launcher ------------------------------------
extern "C" void kernel_launch(void* const* d_in, const int* in_sizes, int n_in,
                              void* d_out, int out_size, void* d_ws, size_t ws_size,
                              hipStream_t stream) {
  const float* x         = (const float*)d_in[0];
  const float* fc1_w     = (const float*)d_in[2];
  const float* fc1_b     = (const float*)d_in[3];
  const float* conv_w    = (const float*)d_in[4];
  const float* conv_b    = (const float*)d_in[5];
  const float* fc2_w     = (const float*)d_in[6];
  const float* fc2_b     = (const float*)d_in[7];
  const float* off_fc1_w = (const float*)d_in[8];
  const float* off_fc1_b = (const float*)d_in[9];
  const float* off_conv_w= (const float*)d_in[10];
  const float* off_conv_b= (const float*)d_in[11];
  const float* off_fc2_w = (const float*)d_in[12];
  const float* off_fc2_b = (const float*)d_in[13];
  const float* mlp_in_w  = (const float*)d_in[14];
  const float* mlp_in_b  = (const float*)d_in[15];
  const float* mlp_out_w = (const float*)d_in[16];
  const float* mlp_out_b = (const float*)d_in[17];
  float* out = (float*)d_out;

  unsigned short* W1t  = (unsigned short*)d_ws;                 // 576*768
  unsigned short* W2t  = W1t  + (size_t)N1 * K1;                // 768*576
  unsigned short* Xb   = W2t  + (size_t)N2 * K2;                // 50432*768
  unsigned short* Utmp = Xb   + (size_t)Mtot * Cc;              // 50432*384
  unsigned short* U2   = Utmp + (size_t)Mtot * 384;             // 50432*576

  convert_x<<<(Mtot * Cc) / (256 * 8), 256, 0, stream>>>(x, Xb);

  prep_weights<<<(2 * N1 * K1 + 255) / 256, 256, 0, stream>>>(
      fc1_w, mlp_in_w, off_fc1_w, fc2_w, mlp_out_w, off_fc2_w, W1t, W2t);

  gemm1_wmma<<<dim3(N1 / 64, Mtot / 256), 256, 0, stream>>>(
      Xb, W1t, fc1_b, mlp_in_b, Utmp, U2);

  stage2_conv<<<(Mtot * CA + 255) / 256, 256, 0, stream>>>(
      Utmp, conv_w, conv_b, off_fc1_b, off_conv_w, off_conv_b, U2);

  gemm2_wmma<<<dim3(N2 / 64, Mtot / 256), 256, 0, stream>>>(
      U2, W2t, fc2_b, mlp_out_b, off_fc2_b, out);
}
